// NTM_19756849562210
// MI455X (gfx1250) — matrix operand order repaired and checked by reading, not compiled, not run on hardware
//
#include <hip/hip_runtime.h>

// ---------------------------------------------------------------------------
// NTM (LSTM controller + 1 read/1 write head) persistent kernel for gfx1250.
// Design:
//  - 4 workgroups x 512 threads; each WG owns a 16-row batch tile and loops
//    over all S=64 timesteps (recurrence is batch-independent).
//  - GEMMs use v_wmma_f32_16x16x32_bf16; weights pre-packed into fragment-
//    native bf16 layout so each lane loads its 32B fragment contiguously.
//  - mem (B x 1024 x 64 f32 = 16MB) lives in d_ws -> fully L2-resident.
//  - addressing / softmax / read / erase-add update: wave == batch row,
//    wave32 shfl reductions, float4/float2 vector memory ops.
// ---------------------------------------------------------------------------

#define EPSF 1e-8f

typedef __bf16 v16bf __attribute__((ext_vector_type(16)));
typedef float  v8f   __attribute__((ext_vector_type(8)));

static constexpr int Bsz = 64, Ssz = 64, Isz = 64, Osz = 64;
static constexpr int Nmem = 1024, Mmem = 64, Hsz = 256;
static constexpr int KC  = 384;              // I + M + H  (A = [x | r | h])
static constexpr int NG  = 1024;             // 4*H gate columns
static constexpr int PP  = 272;              // P=268 padded to 17 col tiles
static constexpr int PR  = 268;
static constexpr int KT_C = 12, NT_C = 64;   // gates GEMM tiles (K=384,N=1024)
static constexpr int KT_H = 8,  NT_H = 17;   // head GEMM tiles (K=256,N=272)
static constexpr int KT_O = 10, NT_O = 4;    // out  GEMM tiles (K=320,N=64)

// ---- LDS layout (dynamic, bytes) ----
static constexpr int OFF_GATES = 0;               // 16x1024 f32 (aliases wrbuf)
static constexpr int OFF_WW    = 65536;           // 16x1024 f32 (wwbuf)
static constexpr int OFF_ABF   = 131072;          // 16x384 bf16
static constexpr int OFF_P     = 143360;          // 16x272 f32
static constexpr int OFF_C     = 160768;          // 16x256 f32
static constexpr int OFF_KR    = 177152;          // 16x64  f32
static constexpr int OFF_KW    = 181248;          // 16x64  f32
static constexpr int OFF_E     = 185344;          // 16x64  f32
static constexpr int OFF_A     = 189440;          // 16x64  f32
static constexpr int OFF_SC    = 193536;          // 16x16  f32 scalars
static constexpr int SMEM_BYTES = 194560;

__device__ __forceinline__ unsigned short f2bf(float f) {
  unsigned u = __builtin_bit_cast(unsigned, f);
  u += 0x7fffu + ((u >> 16) & 1u);               // round-to-nearest-even
  return (unsigned short)(u >> 16);
}
__device__ __forceinline__ float sigmf(float x)     { return 1.f / (1.f + __expf(-x)); }
__device__ __forceinline__ float softplusf(float x) { return (x > 20.f) ? x : __logf(1.f + __expf(x)); }
__device__ __forceinline__ float wave_max(float v)  { for (int o=16;o;o>>=1) v = fmaxf(v, __shfl_xor(v,o,32)); return v; }
__device__ __forceinline__ float wave_sum(float v)  { for (int o=16;o;o>>=1) v += __shfl_xor(v,o,32); return v; }

// 16-bit A-matrix 16x32 layout: lanes 0-15 / 16-31 hold M=0..15, element j
// holds K = half*16 + j (contiguous, matching B packing).
__device__ __forceinline__ v16bf build_a(const unsigned short* Abf, int lm, int half, int kofs) {
  v16bf a;
#pragma unroll
  for (int j = 0; j < 16; ++j) {
    int vi = j >> 1;
    int k  = half * 16 + j; (void)vi;
    a[j] = __builtin_bit_cast(__bf16, Abf[lm * KC + kofs + k]);
  }
  return a;
}

__device__ __forceinline__ v8f wmma_bf16(v16bf a, v16bf b, v8f c) {
  return __builtin_amdgcn_wmma_f32_16x16x32_bf16(false, a, false, b, (short)0, c, false, false);
}

// ---------------------------------------------------------------------------
// Init kernel: mem = 0.01, w_r/w_w one-hot at n==0.
// ---------------------------------------------------------------------------
__global__ void ntm_init(float* mem, float* wrG, float* wwG) {
  size_t g = (size_t)blockIdx.x * blockDim.x + threadIdx.x;
  const size_t NM = (size_t)Bsz * Nmem * Mmem;   // 16M
  if (g < NM) { mem[g] = 0.01f; return; }
  size_t k = g - NM;
  if (k < (size_t)Bsz * Nmem)       { wrG[k] = ((k & (Nmem-1)) == 0) ? 1.f : 0.f; }
  else {
    k -= (size_t)Bsz * Nmem;
    if (k < (size_t)Bsz * Nmem)     { wwG[k] = ((k & (Nmem-1)) == 0) ? 1.f : 0.f; }
  }
}

// ---------------------------------------------------------------------------
// Weight conversion: f32 -> bf16 fragment-native packing.
// Fragment f holds 512 bf16: element = f*512 + lane*16 + j, where
// n = tile_n*16 + (lane&15),  k = tile_k*32 + (lane>=16 ? 16:0) + j.
// Wc  : fused [Wx ; Wh] (384 x 1024), f = ct*12 + kt
// Whd : W_head padded   (256 x 272),  f = ct*8  + kt
// Wo  : W_out row-permuted to [r|h] A ordering (320 x 64), f = ct*10 + kt
// ---------------------------------------------------------------------------
__global__ void ntm_convert(const float* Wx, const float* Wh, const float* Whs,
                            const float* Wos, unsigned short* Wc,
                            unsigned short* Whd, unsigned short* Wo) {
  int gid = blockIdx.x * blockDim.x + threadIdx.x;
  const int NCc = NT_C * KT_C * 512, NCh = NT_H * KT_H * 512, NCo = NT_O * KT_O * 512;
  if (gid < NCc) {
    int f = gid >> 9, r = gid & 511, lanei = r >> 4, j = r & 15;
    int tn = f / KT_C, tk = f % KT_C;
    int n = tn * 16 + (lanei & 15);
    int k = tk * 32 + ((lanei >> 4) << 4) + j;
    float v = (k < 128) ? Wx[k * NG + n] : Wh[(k - 128) * NG + n];
    Wc[gid] = f2bf(v);
  } else if (gid < NCc + NCh) {
    int idx = gid - NCc;
    int f = idx >> 9, r = idx & 511, lanei = r >> 4, j = r & 15;
    int tn = f / KT_H, tk = f % KT_H;
    int n = tn * 16 + (lanei & 15);
    int k = tk * 32 + ((lanei >> 4) << 4) + j;
    float v = (n < PR) ? Whs[k * PR + n] : 0.f;
    Whd[idx] = f2bf(v);
  } else if (gid < NCc + NCh + NCo) {
    int idx = gid - NCc - NCh;
    int f = idx >> 9, r = idx & 511, lanei = r >> 4, j = r & 15;
    int tn = f / KT_O, tk = f % KT_O;
    int n  = tn * 16 + (lanei & 15);
    int kp = tk * 32 + ((lanei >> 4) << 4) + j;     // 0..63 = r rows, 64..319 = h rows
    int k  = (kp < 64) ? (256 + kp) : (kp - 64);
    Wo[idx] = f2bf(Wos[k * Osz + n]);
  }
}

// ---------------------------------------------------------------------------
// Per-head addressing finish: softmax -> interpolate -> shift -> sharpen.
// wbuf[m][n] holds z = beta*sim on entry, final w on exit (also -> wglob).
// Wave == batch row m; uniform across all 16 waves (internal barriers OK).
// ---------------------------------------------------------------------------
__device__ void process_head(float* wbuf, const float* sc, float* wglob,
                             int m, int lane, int bbase) {
  float g = sc[1], s0 = sc[2], s1 = sc[3], s2 = sc[4], gamma = sc[5];
  float* wm = wbuf + m * Nmem;
  float mx = -1e30f;
  for (int ii = 0; ii < 32; ++ii) mx = fmaxf(mx, wm[ii * 32 + lane]);
  mx = wave_max(mx);
  float sum = 0.f;
  for (int ii = 0; ii < 32; ++ii) {
    int n = ii * 32 + lane;
    float e = __expf(wm[n] - mx);
    wm[n] = e; sum += e;
  }
  sum = wave_sum(sum);
  float inv = 1.f / sum;
  const float* wprev = wglob + (size_t)(bbase + m) * Nmem;
  for (int ii = 0; ii < 32; ++ii) {
    int n = ii * 32 + lane;
    wm[n] = g * (wm[n] * inv) + (1.f - g) * wprev[n];     // interpolate
  }
  __syncthreads();
  float wp[32]; float psum = 0.f;
#pragma unroll
  for (int ii = 0; ii < 32; ++ii) {
    int n = ii * 32 + lane;
    float ws = s0 * wm[(n + 1) & (Nmem-1)] + s1 * wm[n] + s2 * wm[(n - 1) & (Nmem-1)];
    float v = __powf(fmaxf(ws, 0.f), gamma);              // sharpen
    wp[ii] = v; psum += v;
  }
  psum = wave_sum(psum);
  float rinv = 1.f / (psum + EPSF);
  __syncthreads();
  float* wout = wglob + (size_t)(bbase + m) * Nmem;
#pragma unroll
  for (int ii = 0; ii < 32; ++ii) {
    int n = ii * 32 + lane;
    float wf = wp[ii] * rinv;
    wm[n] = wf; wout[n] = wf;
  }
}

// ---------------------------------------------------------------------------
// Main persistent kernel: grid 4 x 512 threads, loops t = 0..63.
// ---------------------------------------------------------------------------
__global__ __launch_bounds__(512, 1)
void ntm_main(const float* __restrict__ xin,
              const float* __restrict__ bL, const float* __restrict__ bH,
              const float* __restrict__ bO,
              const unsigned short* __restrict__ Wc,
              const unsigned short* __restrict__ Whd,
              const unsigned short* __restrict__ Wo,
              float* __restrict__ mem, float* __restrict__ wrG,
              float* __restrict__ wwG, float* __restrict__ out) {
  extern __shared__ unsigned char smem[];
  float*          gatesL = (float*)(smem + OFF_GATES);   // also wrbuf
  float*          wrbuf  = gatesL;
  float*          wwbuf  = (float*)(smem + OFF_WW);
  unsigned short* Abf    = (unsigned short*)(smem + OFF_ABF);
  float*          pL     = (float*)(smem + OFF_P);
  float*          cL     = (float*)(smem + OFF_C);
  float*          krL    = (float*)(smem + OFF_KR);
  float*          kwL    = (float*)(smem + OFF_KW);
  float*          eL     = (float*)(smem + OFF_E);
  float*          aL     = (float*)(smem + OFF_A);
  float*          scal   = (float*)(smem + OFF_SC);

  const int tid = threadIdx.x, lane = tid & 31, wave = tid >> 5;
  const int bbase = blockIdx.x * 16;
  const int half = lane >> 4, lm = lane & 15;

  for (int i = tid; i < 16 * Hsz; i += 512) cL[i] = 0.f;
  for (int i = tid; i < 16 * KC;  i += 512) Abf[i] = 0;   // r=0, h=0
  __syncthreads();

  for (int t = 0; t < Ssz; ++t) {
    // ---- stage 0: stage x_t into A ----
    for (int i = tid; i < 16 * Isz; i += 512) {
      int m = i >> 6, col = i & 63;
      Abf[m * KC + col] = f2bf(xin[((size_t)(bbase + m) * Ssz + t) * Isz + col]);
    }
    __syncthreads();

    // ---- stage 1: gates = [x|r|h] @ [Wx;Wh] + b  (WMMA bf16) ----
    for (int q = 0; q < 4; ++q) {
      int ct = wave + (q << 4);
      v8f acc = {0.f,0.f,0.f,0.f,0.f,0.f,0.f,0.f};
#pragma unroll
      for (int kt = 0; kt < KT_C; ++kt) {
        v16bf a = build_a(Abf, lm, half, kt * 32);
        v16bf b = *(const v16bf*)(Wc + (((size_t)ct * KT_C + kt) << 9) + (lane << 4));
        acc = wmma_bf16(a, b, acc);
      }
      int n = (ct << 4) + lm;
#pragma unroll
      for (int i = 0; i < 8; ++i)
        gatesL[(i + (half << 3)) * NG + n] = acc[i] + bL[n];
    }
    __syncthreads();

    // ---- stage 2: LSTM elementwise; h -> A (bf16) ----
    for (int i = tid; i < 16 * Hsz; i += 512) {
      int m = i >> 8, j = i & 255;
      float ig = gatesL[m * NG + j],        fg = gatesL[m * NG + 256 + j];
      float gg = gatesL[m * NG + 512 + j],  og = gatesL[m * NG + 768 + j];
      float c = sigmf(fg) * cL[i] + sigmf(ig) * tanhf(gg);
      float h = sigmf(og) * tanhf(c);
      cL[i] = c;
      Abf[m * KC + 128 + j] = f2bf(h);
    }
    __syncthreads();

    // ---- stage 3: p = h @ W_head + b_head ----
    {
      int nct = (wave == 0) ? 2 : 1;              // wave 0 takes tiles 0 and 16
      for (int q = 0; q < nct; ++q) {
        int ct = wave + (q << 4);
        v8f acc = {0.f,0.f,0.f,0.f,0.f,0.f,0.f,0.f};
#pragma unroll
        for (int kt = 0; kt < KT_H; ++kt) {
          v16bf a = build_a(Abf, lm, half, 128 + kt * 32);
          v16bf b = *(const v16bf*)(Whd + (((size_t)ct * KT_H + kt) << 9) + (lane << 4));
          acc = wmma_bf16(a, b, acc);
        }
        int n = (ct << 4) + lm;
        float bias = (n < PR) ? bH[n] : 0.f;
#pragma unroll
        for (int i = 0; i < 8; ++i)
          pL[(i + (half << 3)) * PP + n] = acc[i] + bias;
      }
    }
    __syncthreads();

    // ---- stage 4: parse heads (wave m <-> batch m) ----
    {
      int m = wave;
      const float* pm = pL + m * PP;
      float nr = 0.f, nw = 0.f;
      for (int rep = 0; rep < 2; ++rep) {
        int j = lane + rep * 32;
        float kr = tanhf(pm[j]);        krL[m * 64 + j] = kr; nr += kr * kr;
        float kw = tanhf(pm[70 + j]);   kwL[m * 64 + j] = kw; nw += kw * kw;
        eL[m * 64 + j] = sigmf(pm[140 + j]);
        aL[m * 64 + j] = pm[204 + j];
      }
      nr = wave_sum(nr); nw = wave_sum(nw);
      if (lane == 0) {
        float* sc = scal + m * 16;
        // read head: k|beta|g|s0..2|gamma at p[0..69]
        sc[0] = softplusf(pm[64]); sc[1] = sigmf(pm[65]);
        { float a0=pm[66], a1=pm[67], a2=pm[68];
          float mx = fmaxf(a0, fmaxf(a1, a2));
          float e0=__expf(a0-mx), e1=__expf(a1-mx), e2=__expf(a2-mx);
          float s = e0+e1+e2; sc[2]=e0/s; sc[3]=e1/s; sc[4]=e2/s; }
        sc[5] = 1.f + softplusf(pm[69]); sc[6] = sqrtf(nr);
        // write head: p[70..139]
        sc[8] = softplusf(pm[134]); sc[9] = sigmf(pm[135]);
        { float a0=pm[136], a1=pm[137], a2=pm[138];
          float mx = fmaxf(a0, fmaxf(a1, a2));
          float e0=__expf(a0-mx), e1=__expf(a1-mx), e2=__expf(a2-mx);
          float s = e0+e1+e2; sc[10]=e0/s; sc[11]=e1/s; sc[12]=e2/s; }
        sc[13] = 1.f + softplusf(pm[139]); sc[14] = sqrtf(nw);
      }
    }
    __syncthreads();

    // ---- stage 5: content similarity (single mem pass, both heads) ----
    {
      int m = wave;
      const float* sc = scal + m * 16;
      float knr = sc[6], knw = sc[14], br = sc[0], bw = sc[8];
      const float4* kr4 = (const float4*)(krL + m * 64);
      const float4* kw4 = (const float4*)(kwL + m * 64);
      for (int ii = 0; ii < 32; ++ii) {
        int n = ii * 32 + lane;
        const float4* row = (const float4*)(mem + ((size_t)(bbase + m) * Nmem + n) * Mmem);
        float dr = 0.f, dw = 0.f, nm = 0.f;
#pragma unroll 4
        for (int j = 0; j < 16; ++j) {
          float4 v = row[j], kr = kr4[j], kw = kw4[j];
          dr += v.x*kr.x + v.y*kr.y + v.z*kr.z + v.w*kr.w;
          dw += v.x*kw.x + v.y*kw.y + v.z*kw.z + v.w*kw.w;
          nm += v.x*v.x  + v.y*v.y  + v.z*v.z  + v.w*v.w;
        }
        float nrm = sqrtf(nm);
        wrbuf[m * Nmem + n] = br * (dr / (knr * nrm + EPSF));
        wwbuf[m * Nmem + n] = bw * (dw / (knw * nrm + EPSF));
      }
      process_head(wrbuf, sc,     wrG, m, lane, bbase);    // read head
      process_head(wwbuf, sc + 8, wwG, m, lane, bbase);    // write head
    }
    __syncthreads();

    // ---- stage 6: r = w_r . mem (old), then erase/add update; r -> A ----
    {
      int m = wave;
      int col = lane << 1;                                  // 2 cols per lane, coalesced
      float e0 = eL[m * 64 + col], e1 = eL[m * 64 + col + 1];
      float a0 = aL[m * 64 + col], a1 = aL[m * 64 + col + 1];
      float r0 = 0.f, r1 = 0.f;
      float* base = mem + (size_t)(bbase + m) * Nmem * Mmem + col;
      for (int n = 0; n < Nmem; ++n) {
        float wr = wrbuf[m * Nmem + n], ww = wwbuf[m * Nmem + n];
        float2 v = *(float2*)(base + (size_t)n * Mmem);
        r0 += wr * v.x; r1 += wr * v.y;
        v.x = v.x * (1.f - ww * e0) + ww * a0;
        v.y = v.y * (1.f - ww * e1) + ww * a1;
        *(float2*)(base + (size_t)n * Mmem) = v;
      }
      Abf[m * KC + 64 + col]     = f2bf(r0);
      Abf[m * KC + 64 + col + 1] = f2bf(r1);
    }
    __syncthreads();

    // ---- stage 7: out = [r|h] @ W_out_perm + b_out ----
    if (wave < NT_O) {
      int ct = wave;
      v8f acc = {0.f,0.f,0.f,0.f,0.f,0.f,0.f,0.f};
#pragma unroll
      for (int kt = 0; kt < KT_O; ++kt) {
        v16bf a = build_a(Abf, lm, half, 64 + kt * 32);
        v16bf b = *(const v16bf*)(Wo + (((size_t)ct * KT_O + kt) << 9) + (lane << 4));
        acc = wmma_bf16(a, b, acc);
      }
      int n = (ct << 4) + lm;
#pragma unroll
      for (int i = 0; i < 8; ++i) {
        int mrow = i + (half << 3);
        out[((size_t)(bbase + mrow) * Ssz + t) * Osz + n] = acc[i] + bO[n];
      }
    }
    __syncthreads();
  }
}

// ---------------------------------------------------------------------------
extern "C" void kernel_launch(void* const* d_in, const int* in_sizes, int n_in,
                              void* d_out, int out_size, void* d_ws, size_t ws_size,
                              hipStream_t stream) {
  const float* x    = (const float*)d_in[0];   // (64,64,64)
  const float* Wx   = (const float*)d_in[1];   // (128,1024)
  const float* Wh   = (const float*)d_in[2];   // (256,1024)
  const float* bL   = (const float*)d_in[3];   // (1024,)
  const float* Whs  = (const float*)d_in[4];   // (256,268)
  const float* bH   = (const float*)d_in[5];   // (268,)
  const float* Wos  = (const float*)d_in[6];   // (320,64)
  const float* bO   = (const float*)d_in[7];   // (64,)
  float* out = (float*)d_out;

  // scratch carve (bytes)
  unsigned char* ws = (unsigned char*)d_ws;
  float* mem = (float*)ws;                                   // 16 MB
  float* wrG = (float*)(ws + 16777216);                      // 256 KB
  float* wwG = (float*)(ws + 16777216 + 262144);             // 256 KB
  unsigned short* Wc  = (unsigned short*)(ws + 17301504);    // 768 KB
  unsigned short* Whd = (unsigned short*)(ws + 18087936);    // 136 KB
  unsigned short* Wo  = (unsigned short*)(ws + 18227200);    // 40 KB

  // state init: 16M mem + 2x64K w-vectors
  ntm_init<<<66048, 256, 0, stream>>>(mem, wrG, wwG);
  // weight pack: (768+136+40)*512 elements
  ntm_convert<<<1888, 256, 0, stream>>>(Wx, Wh, Whs, Wos, Wc, Whd, Wo);

  (void)hipFuncSetAttribute((const void*)ntm_main,
                            hipFuncAttributeMaxDynamicSharedMemorySize, SMEM_BYTES);
  ntm_main<<<4, 512, SMEM_BYTES, stream>>>(x, bL, bH, bO, Wc, Whd, Wo,
                                           mem, wrG, wwG, out);
  (void)in_sizes; (void)n_in; (void)out_size; (void)ws_size;
}
